// BiLSTMTagger_8744553415089
// MI455X (gfx1250) — matrix operand order repaired and checked by reading, not compiled, not run on hardware
//
#include <hip/hip_runtime.h>
#include <hip/hip_bf16.h>

// BiLSTM tagger for MI455X (gfx1250, wave32, WMMA).
// bf16 operands + f32 accumulate via v_wmma_f32_16x16x32_bf16.
// - GEMMs: LDS double-buffered B slabs (ds_load_b128 feeding WMMA).
// - Recurrent scan: persistent kernel, Whh slab resident in LDS (loaded once
//   via TDM tensor_load_to_lds when available), agent-scope grid barrier.

#define B_    128
#define T_    512
#define E_    256
#define H_    512
#define G4_   2048
#define TAGS_ 64
#define NSCANWG 64   // persistent scan workgroups (x 256 threads = 512 waves)

typedef __attribute__((ext_vector_type(16))) __bf16 v16bf;
typedef __attribute__((ext_vector_type(8)))  float  v8f;

union FragU { v16bf v; unsigned u[8]; };

#if defined(__has_builtin)
#if __has_builtin(__builtin_amdgcn_tensor_load_to_lds) && \
    __has_builtin(__builtin_amdgcn_s_wait_tensorcnt)
#define HAS_TDM 1
#endif
#endif
#ifndef HAS_TDM
#define HAS_TDM 0
#endif

// ---- WMMA fragment loaders (ISA 7.12.2 layouts) --------------------------

// A matrix 16x32 bf16 (MxK), row-major source, ld in elements.
// lane l: row = l&15 ; K = {0..7,16..23} + 8*(l>=16), packed 2/dword.
__device__ __forceinline__ v16bf load_a_frag(const __hip_bfloat16* X, int ld,
                                             int row0, int k0) {
  const int lane = threadIdx.x & 31;
  const int row  = row0 + (lane & 15);
  const unsigned* p = reinterpret_cast<const unsigned*>(X + (size_t)row * ld);
  const int base = (k0 >> 1) + ((lane >> 4) << 2);   // dword index
  FragU f;
#pragma unroll
  for (int i = 0; i < 4; ++i) {
    f.u[i]     = p[base + i];       // K = kb+0..7
    f.u[4 + i] = p[base + 8 + i];   // K = kb+16..23
  }
  return f.v;
}

// B matrix 32x16 bf16 (KxN) taken from row-major weight W[N,K] (so column j
// of B == row j of W, contiguous in K).  lane l: col = l&15 ;
// K = {0..7,16..23} + 8*(l>=16), packed 2/dword.
// Works for global or LDS (__shared__) source pointers.
__device__ __forceinline__ v16bf load_b_frag(const __hip_bfloat16* W, int ld,
                                             int col0, int k0) {
  const int lane = threadIdx.x & 31;
  const int col  = col0 + (lane & 15);
  const unsigned* p = reinterpret_cast<const unsigned*>(W + (size_t)col * ld);
  const int base = (k0 >> 1) + ((lane >> 4) << 2);   // dword index
  FragU f;
#pragma unroll
  for (int i = 0; i < 4; ++i) {
    f.u[i]     = p[base + i];
    f.u[4 + i] = p[base + 8 + i];
  }
  return f.v;
}

__device__ __forceinline__ v8f wmma_bf16(v16bf a, v16bf b, v8f c) {
  return __builtin_amdgcn_wmma_f32_16x16x32_bf16(false, a, false, b,
                                                 (short)0, c, false, false);
}

__device__ __forceinline__ float sigmoidf(float x) {
  return 1.0f / (1.0f + __expf(-x));
}

// ---- TDM: contiguous 32KB global -> LDS (1-D descriptor) -----------------
#if HAS_TDM
typedef unsigned u32x4 __attribute__((ext_vector_type(4)));
typedef int      i32x4 __attribute__((ext_vector_type(4)));
typedef int      i32x8 __attribute__((ext_vector_type(8)));

__device__ __forceinline__ void tdm_load_32kb(unsigned lds_byte_off,
                                              const void* gsrc) {
  unsigned long long ga = (unsigned long long)gsrc;
  u32x4 g0;
  g0.x = 1u;                                     // count=1, user descriptor
  g0.y = lds_byte_off;                           // lds_addr (bytes)
  g0.z = (unsigned)(ga & 0xffffffffu);           // global_addr[31:0]
  g0.w = (unsigned)((ga >> 32) & 0x1ffffffu) | (2u << 30);  // [56:32]|type=2
  i32x8 g1;
  g1[0] = (3 << 16);          // workgroup_mask=0, data_size=3 (8 bytes)
  g1[1] = 0x10000000;         // tensor_dim0 = 4096  (bits 63:48 of value)
  g1[2] = 0;                  // tensor_dim0 hi / tensor_dim1 = 0
  g1[3] = 0x10000000;         // tile_dim0 = 4096 (bits 127:112)
  g1[4] = 0;                  // tile_dim1 = tile_dim2 = 0 (1-D)
  g1[5] = 4096;               // tensor_dim0_stride (unused for 1 row)
  g1[6] = 0;
  g1[7] = 0;
  i32x4 z4 = {0, 0, 0, 0};                       // groups 2/3: NULL (<=2D)
  i32x8 z8 = {0, 0, 0, 0, 0, 0, 0, 0};
  __builtin_amdgcn_tensor_load_to_lds(g0, g1, z4, z4, z8, 0);
}
#endif

// ---- agent-scope grid barrier (persistent scan kernel) -------------------
__device__ __forceinline__ void grid_barrier(unsigned* cnt, unsigned* gen) {
  __syncthreads();
  if (threadIdx.x == 0) {
    unsigned g = __hip_atomic_load(gen, __ATOMIC_ACQUIRE, __HIP_MEMORY_SCOPE_AGENT);
    unsigned a = __hip_atomic_fetch_add(cnt, 1u, __ATOMIC_ACQ_REL, __HIP_MEMORY_SCOPE_AGENT);
    if (a == gridDim.x - 1) {
      __hip_atomic_store(cnt, 0u, __ATOMIC_RELAXED, __HIP_MEMORY_SCOPE_AGENT);
      __hip_atomic_fetch_add(gen, 1u, __ATOMIC_ACQ_REL, __HIP_MEMORY_SCOPE_AGENT);
    } else {
      while (__hip_atomic_load(gen, __ATOMIC_ACQUIRE, __HIP_MEMORY_SCOPE_AGENT) == g) {
        __builtin_amdgcn_s_sleep(2);
      }
    }
  }
  __syncthreads();
}

// ---- small elementwise kernels -------------------------------------------

__global__ void k_cast_bf16(const float* __restrict__ in,
                            __hip_bfloat16* __restrict__ out, int n) {
  int i = blockIdx.x * blockDim.x + threadIdx.x;
  if (i < n) out[i] = __float2bfloat16(in[i]);
}

__global__ void k_embed(const int* __restrict__ x, const float* __restrict__ emb,
                        __hip_bfloat16* __restrict__ e, int total) {
  int i = blockIdx.x * blockDim.x + threadIdx.x;
  if (i >= total) return;
  int row = i >> 8;            // /E_
  int col = i & (E_ - 1);
  e[i] = __float2bfloat16(emb[(size_t)x[row] * E_ + col]);
}

// out[b,t, outOff + d] = (len[b]-1-t >= 0) ? in[b, len[b]-1-t, d] : 0
__global__ void k_reverse(const __hip_bfloat16* __restrict__ in,
                          __hip_bfloat16* __restrict__ out,
                          const int* __restrict__ len,
                          int D, int inLd, int outLd, int outOff, int total) {
  int i = blockIdx.x * blockDim.x + threadIdx.x;
  if (i >= total) return;
  int d  = i % D;
  int bt = i / D;
  int t  = bt & (T_ - 1);
  int b  = bt >> 9;            // /T_
  int idx = len[b] - 1 - t;
  __hip_bfloat16 v = (idx >= 0)
      ? in[((size_t)b * T_ + idx) * inLd + d]
      : __float2bfloat16(0.0f);
  out[((size_t)b * T_ + t) * outLd + outOff + d] = v;
}

// ---- WMMA GEMM:  C[M,N](f32) = A[M,K](bf16) * W[N,K](bf16)^T + bias[N] ---
// block = 4 waves, 64x64 tile.  B slab double-buffered through LDS; each
// wave: 16 rows x 64 cols (A reuse x4, B shared by all 4 waves via LDS).
__global__ __launch_bounds__(128)
void k_gemm_bias(const __hip_bfloat16* __restrict__ A,
                 const __hip_bfloat16* __restrict__ W,
                 const float* __restrict__ bias,
                 float* __restrict__ C, int M, int N, int K) {
  __shared__ __hip_bfloat16 Bs[2][64 * 32];     // 2 x 4KB
  const int wave = threadIdx.x >> 5;
  const int lane = threadIdx.x & 31;
  const int row0 = blockIdx.x * 64 + wave * 16;
  const int col0 = blockIdx.y * 64;
  const int scol  = threadIdx.x >> 1;           // 0..63: B row (=output col)
  const int skoff = (threadIdx.x & 1) * 16;     // elem offset within 32-K slab
  const int nch = K >> 5;

  uint4 s0, s1;
  {
    const uint4* p = (const uint4*)(W + (size_t)(col0 + scol) * K + skoff);
    s0 = p[0]; s1 = p[1];
    uint4* d = (uint4*)(Bs[0] + scol * 32 + skoff);
    d[0] = s0; d[1] = s1;
  }
  __syncthreads();

  v8f acc[4] = {};
  for (int ci = 0; ci < nch; ++ci) {
    if (ci + 1 < nch) {   // prefetch next B slab into registers
      const uint4* p = (const uint4*)(W + (size_t)(col0 + scol) * K +
                                      (ci + 1) * 32 + skoff);
      s0 = p[0]; s1 = p[1];
    }
    v16bf a = load_a_frag(A, K, row0, ci * 32);
    const __hip_bfloat16* Bb = Bs[ci & 1];
#pragma unroll
    for (int j = 0; j < 4; ++j) {
      v16bf b = load_b_frag(Bb, 32, 16 * j, 0);
      acc[j] = wmma_bf16(a, b, acc[j]);
    }
    if (ci + 1 < nch) {
      __syncthreads();    // readers of the other buffer (2 chunks ago) done
      uint4* d = (uint4*)(Bs[(ci + 1) & 1] + scol * 32 + skoff);
      d[0] = s0; d[1] = s1;
      __syncthreads();
    }
  }

  const int rsub = (lane >> 4) << 3;
  const int csub = lane & 15;
#pragma unroll
  for (int j = 0; j < 4; ++j) {
    int col = col0 + 16 * j + csub;
    float bv = bias ? bias[col] : 0.0f;
#pragma unroll
    for (int v = 0; v < 8; ++v) {
      int row = row0 + rsub + v;
      C[(size_t)row * N + col] = acc[j][v] + bv;
    }
  }
}

// ---- persistent LSTM scan -------------------------------------------------
// grid = NSCANWG(=64) blocks x 256 threads = 512 waves (all resident).
// block b owns output columns [32b, 32b+32); its Whh slab (32 rows x 512 K,
// 32KB contiguous) is loaded into LDS ONCE (TDM if available) and reused for
// all 512 time steps.  Wave w computes M-tile w (rows 16w..16w+15).
// per step t:
//   phase1: g[b,:] = gx[b,t,:] + h @ Whh^T   (A from global h, B from LDS)
//   phase2: gates -> c,h ; h written (masked) to out[b,t,outOff + :H]
__global__ __launch_bounds__(256)
void k_lstm_scan(const float* __restrict__ gx,            // [B,T,4H]
                 const __hip_bfloat16* __restrict__ Whh,  // [4H,H] bf16
                 float* __restrict__ gs,                  // [B,4H] scratch
                 __hip_bfloat16* __restrict__ h,          // [B,H] bf16 state
                 float* __restrict__ c,                   // [B,H] f32 state
                 __hip_bfloat16* __restrict__ out,        // [B,T,outLd]
                 const int* __restrict__ lengths,         // null => unmasked
                 int outLd, int outOff,
                 unsigned* bar_cnt, unsigned* bar_gen) {
  __shared__ __hip_bfloat16 Wslab[32 * 512];   // 32KB, the ONLY shared object
  const int wave    = threadIdx.x >> 5;        // 0..7 == M-tile
  const int lane    = threadIdx.x & 31;
  const int mrow0   = wave << 4;
  const int colBase = blockIdx.x << 5;         // 0,32,..,2016
  const int rsub    = (lane >> 4) << 3;
  const int csub    = lane & 15;
  const int tid     = blockIdx.x * 256 + threadIdx.x;     // 0..16383

  // ---- one-time Whh slab load into LDS ----
#if HAS_TDM
  if (threadIdx.x < 32) {              // wave 0 issues the DMA
    tdm_load_32kb(0u, Whh + (size_t)colBase * H_);
    __builtin_amdgcn_s_wait_tensorcnt(0);
  }
#else
  {
    int col = threadIdx.x >> 3;        // 0..31
    int seg = threadIdx.x & 7;         // 64 elems (128B) each
    const uint4* src = (const uint4*)(Whh + (size_t)(colBase + col) * H_ + seg * 64);
    uint4* dst = (uint4*)(Wslab + col * H_ + seg * 64);
#pragma unroll
    for (int i = 0; i < 8; ++i) dst[i] = src[i];
  }
#endif
  __syncthreads();

  for (int t = 0; t < T_; ++t) {
    // ---- phase 1: recurrent matmul tile (B from LDS) ------------------
    v8f acc0 = {}, acc1 = {};
    for (int k0 = 0; k0 < H_; k0 += 32) {
      v16bf a  = load_a_frag(h, H_, mrow0, k0);
      v16bf b0 = load_b_frag(Wslab, H_, 0, k0);
      acc0 = wmma_bf16(a, b0, acc0);
      v16bf b1 = load_b_frag(Wslab, H_, 16, k0);
      acc1 = wmma_bf16(a, b1, acc1);
    }
#pragma unroll
    for (int v = 0; v < 8; ++v) {
      int bi = mrow0 + rsub + v;
      const float* gxr = gx + ((size_t)bi * T_ + t) * G4_;
      int cc = colBase + csub;
      gs[(size_t)bi * G4_ + cc]      = acc0[v] + gxr[cc];
      gs[(size_t)bi * G4_ + cc + 16] = acc1[v] + gxr[cc + 16];
    }
    grid_barrier(bar_cnt, bar_gen);

    // ---- phase 2: gates (65536 elems / 16384 threads = 4 each) --------
#pragma unroll
    for (int e = 0; e < 4; ++e) {
      int idx = tid + e * (NSCANWG * 256);
      int bi  = idx >> 9;          // /H_
      int j   = idx & (H_ - 1);
      const float* gr = gs + (size_t)bi * G4_ + j;
      float gi = gr[0], gf = gr[H_], gg = gr[2 * H_], go = gr[3 * H_];
      float cv = sigmoidf(gf) * c[idx] + sigmoidf(gi) * tanhf(gg);
      float hv = sigmoidf(go) * tanhf(cv);
      c[idx] = cv;
      h[idx] = __float2bfloat16(hv);
      float ov = hv;
      if (lengths && t >= lengths[bi]) ov = 0.0f;   // pad_packed mask
      out[((size_t)bi * T_ + t) * outLd + outOff + j] = __float2bfloat16(ov);
    }
    grid_barrier(bar_cnt, bar_gen);   // h/c visible before next phase 1
  }
}

// --------------------------------------------------------------------------

extern "C" void kernel_launch(void* const* d_in, const int* in_sizes, int n_in,
                              void* d_out, int out_size, void* d_ws, size_t ws_size,
                              hipStream_t stream) {
  const int*   x      = (const int*)d_in[0];
  const int*   lens   = (const int*)d_in[1];
  const float* emb    = (const float*)d_in[2];
  const float* Wih_f1 = (const float*)d_in[3];
  const float* Whh_f1 = (const float*)d_in[4];
  const float* b_f1   = (const float*)d_in[5];
  const float* Wih_b1 = (const float*)d_in[6];
  const float* Whh_b1 = (const float*)d_in[7];
  const float* b_b1   = (const float*)d_in[8];
  const float* Wih_f2 = (const float*)d_in[9];
  const float* Whh_f2 = (const float*)d_in[10];
  const float* b_f2   = (const float*)d_in[11];
  const float* Wih_b2 = (const float*)d_in[12];
  const float* Whh_b2 = (const float*)d_in[13];
  const float* b_b2   = (const float*)d_in[14];
  const float* Wcls   = (const float*)d_in[15];
  const float* b_cls  = (const float*)d_in[16];
  float* logits = (float*)d_out;
  (void)in_sizes; (void)n_in; (void)out_size; (void)ws_size;

  // ---- workspace carving (256B aligned) ----
  char* wsp = (char*)d_ws; size_t off = 0;
  auto carve = [&](size_t bytes) -> void* {
    size_t a = (off + 255) & ~(size_t)255; off = a + bytes; return (void*)(wsp + a);
  };
  const size_t M = (size_t)B_ * T_;                 // 65536
  __hip_bfloat16* bWih1f = (__hip_bfloat16*)carve((size_t)G4_ * E_ * 2);
  __hip_bfloat16* bWhh1f = (__hip_bfloat16*)carve((size_t)G4_ * H_ * 2);
  __hip_bfloat16* bWih1b = (__hip_bfloat16*)carve((size_t)G4_ * E_ * 2);
  __hip_bfloat16* bWhh1b = (__hip_bfloat16*)carve((size_t)G4_ * H_ * 2);
  __hip_bfloat16* bWih2f = (__hip_bfloat16*)carve((size_t)G4_ * 2 * H_ * 2);
  __hip_bfloat16* bWhh2f = (__hip_bfloat16*)carve((size_t)G4_ * H_ * 2);
  __hip_bfloat16* bWih2b = (__hip_bfloat16*)carve((size_t)G4_ * 2 * H_ * 2);
  __hip_bfloat16* bWhh2b = (__hip_bfloat16*)carve((size_t)G4_ * H_ * 2);
  __hip_bfloat16* bWcls  = (__hip_bfloat16*)carve((size_t)TAGS_ * 2 * H_ * 2);
  __hip_bfloat16* e    = (__hip_bfloat16*)carve(M * E_ * 2);
  __hip_bfloat16* er   = (__hip_bfloat16*)carve(M * E_ * 2);
  float*          gx   = (float*)carve(M * G4_ * 4);   // reused by all 4 scans
  __hip_bfloat16* out1 = (__hip_bfloat16*)carve(M * 2 * H_ * 2);
  __hip_bfloat16* r1   = (__hip_bfloat16*)carve(M * 2 * H_ * 2);
  __hip_bfloat16* out2 = (__hip_bfloat16*)carve(M * 2 * H_ * 2);
  __hip_bfloat16* tmp  = (__hip_bfloat16*)carve(M * H_ * 2);
  __hip_bfloat16* hst  = (__hip_bfloat16*)carve((size_t)B_ * H_ * 2);
  float*          cst  = (float*)carve((size_t)B_ * H_ * 4);
  float*          gs   = (float*)carve((size_t)B_ * G4_ * 4);
  unsigned*       bar  = (unsigned*)carve(2 * sizeof(unsigned));

  auto cast = [&](const float* src, __hip_bfloat16* dst, int n) {
    k_cast_bf16<<<(n + 255) / 256, 256, 0, stream>>>(src, dst, n);
  };
  auto gemm = [&](const __hip_bfloat16* A, const __hip_bfloat16* W,
                  const float* bias, float* C, int m, int n, int k) {
    dim3 g(m / 64, n / 64);
    k_gemm_bias<<<g, 128, 0, stream>>>(A, W, bias, C, m, n, k);
  };
  auto scan = [&](const float* gxp, const __hip_bfloat16* Whh,
                  __hip_bfloat16* outp, const int* lp, int outLd, int outOff) {
    (void)hipMemsetAsync(hst, 0, (size_t)B_ * H_ * 2, stream);
    (void)hipMemsetAsync(cst, 0, (size_t)B_ * H_ * 4, stream);
    k_lstm_scan<<<NSCANWG, 256, 0, stream>>>(gxp, Whh, gs, hst, cst, outp,
                                             lp, outLd, outOff, bar, bar + 1);
  };
  auto reverse = [&](const __hip_bfloat16* in, __hip_bfloat16* outp,
                     int D, int inLd, int outLd, int outOff) {
    int total = (int)(M * D);
    k_reverse<<<(total + 255) / 256, 256, 0, stream>>>(in, outp, lens, D,
                                                       inLd, outLd, outOff, total);
  };

  (void)hipMemsetAsync(bar, 0, 2 * sizeof(unsigned), stream);

  // weights -> bf16 (stay resident in 192MB L2 thereafter)
  cast(Wih_f1, bWih1f, G4_ * E_);      cast(Whh_f1, bWhh1f, G4_ * H_);
  cast(Wih_b1, bWih1b, G4_ * E_);      cast(Whh_b1, bWhh1b, G4_ * H_);
  cast(Wih_f2, bWih2f, G4_ * 2 * H_);  cast(Whh_f2, bWhh2f, G4_ * H_);
  cast(Wih_b2, bWih2b, G4_ * 2 * H_);  cast(Whh_b2, bWhh2b, G4_ * H_);
  cast(Wcls,  bWcls,  TAGS_ * 2 * H_);

  // embedding + reversed copy
  k_embed<<<((int)(M * E_) + 255) / 256, 256, 0, stream>>>(x, emb, e, (int)(M * E_));
  reverse(e, er, E_, E_, E_, 0);

  // ---- layer 1 ----
  gemm(e,  bWih1f, b_f1, gx, (int)M, G4_, E_);
  scan(gx, bWhh1f, out1, lens, 2 * H_, 0);            // fwd, masked
  gemm(er, bWih1b, b_b1, gx, (int)M, G4_, E_);
  scan(gx, bWhh1b, tmp, nullptr, H_, 0);              // bwd on reversed input
  reverse(tmp, out1, H_, H_, 2 * H_, H_);             // un-reverse + zero pad

  // ---- layer 2 ----
  reverse(out1, r1, 2 * H_, 2 * H_, 2 * H_, 0);
  gemm(out1, bWih2f, b_f2, gx, (int)M, G4_, 2 * H_);
  scan(gx, bWhh2f, out2, lens, 2 * H_, 0);
  gemm(r1, bWih2b, b_b2, gx, (int)M, G4_, 2 * H_);
  scan(gx, bWhh2b, tmp, nullptr, H_, 0);
  reverse(tmp, out2, H_, H_, 2 * H_, H_);

  // ---- classifier ----
  gemm(out2, bWcls, b_cls, logits, (int)M, TAGS_, 2 * H_);
}
